// AdaptiveRegisterStep_71347996721400
// MI455X (gfx1250) — matrix operand order repaired and checked by reading, not compiled, not run on hardware
//
#include <hip/hip_runtime.h>
#include <math.h>

// ---------------- problem constants ----------------
constexpr int CB  = 4;        // batch
constexpr int CT  = 2048;     // time
constexpr int CV  = 1024;     // channels
constexpr int CKA = 256;      // active registers
constexpr int CNB = 32;       // basis
constexpr int CNC = 128;      // dct channels
constexpr int CM  = CB * CT;  // flattened rows
constexpr float EPSF = 1.1920929e-07f;   // np.float32 eps

// ---------------- bf16 / WMMA plumbing ----------------
typedef __attribute__((ext_vector_type(16))) __bf16 v16bf;
typedef __attribute__((ext_vector_type(8)))  float  v8f;

struct alignas(16) US8  { unsigned int w[4]; };   // 8 bf16 halves
struct             US16 { US8 lo, hi; };          // 16 bf16 halves

static __device__ inline unsigned short f2bf(float f) {
  unsigned u = __float_as_uint(f);
  unsigned r = u + 0x7FFFu + ((u >> 16) & 1u);
  return (unsigned short)(r >> 16);
}

static __device__ inline v8f wmma_bf16(v16bf a, v16bf b, v8f c) {
  return __builtin_amdgcn_wmma_f32_16x16x32_bf16(
      /*neg_a=*/false, a, /*neg_b=*/false, b,
      /*c_mod=*/(short)0, c, /*reuse_a=*/false, /*reuse_b=*/false);
}

// A fragment (16x32 bf16, M x K), source row-major [16, ld] starting at `base`.
// Lane layout (ISA 7.12.2): lanes 0-15 hold K=k0..k0+7 and k0+16..k0+23 for row M=lane;
// lanes 16-31 hold K=k0+8..k0+15 and k0+24..k0+31 for row M=lane-16.
static __device__ inline v16bf load_a_frag(const unsigned short* base, int ld,
                                           int k0, int lane) {
  int m    = lane & 15;
  int koff = (lane & 16) ? 8 : 0;
  const unsigned short* p = base + (size_t)m * ld + k0 + koff;
  US16 u;
  u.lo = *(const US8*)(p);
  u.hi = *(const US8*)(p + 16);
  return __builtin_bit_cast(v16bf, u);
}

// B fragment (32x16 bf16, K x N): lane n<16 holds B[k0..k0+7][n] and B[k0+16..k0+23][n], lane n+16 holds
// B[k0+8..k0+15][n] and B[k0+24..k0+31][n].  Caller supplies `rowptr` = contiguous K-major storage of
// column n (i.e. weights stored [N, K] row-major, row = this lane's column).
static __device__ inline v16bf load_b_frag(const unsigned short* rowptr,
                                           int k0, int lane) {
  int koff = (lane & 16) ? 8 : 0;
  const unsigned short* p = rowptr + k0 + koff;
  US16 u;
  u.lo = *(const US8*)(p);
  u.hi = *(const US8*)(p + 16);
  return __builtin_bit_cast(v16bf, u);
}

// ---------------- small prep kernels ----------------

// inv-rms per row of a [M, V] fp32 matrix
__global__ void rms_rowinv_kernel(const float* __restrict__ x,
                                  float* __restrict__ rinv, int V) {
  __shared__ float red[256];
  int m = blockIdx.x;
  const float* row = x + (size_t)m * V;
  float s = 0.f;
  for (int v = threadIdx.x; v < V; v += 256) { float t = row[v]; s += t * t; }
  red[threadIdx.x] = s; __syncthreads();
  for (int k = 128; k > 0; k >>= 1) {
    if (threadIdx.x < (unsigned)k) red[threadIdx.x] += red[threadIdx.x + k];
    __syncthreads();
  }
  if (threadIdx.x == 0) rinv[m] = rsqrtf(red[0] / (float)V + EPSF);
}

// normalize row and emit bf16
__global__ void rms_norm_bf16_kernel(const float* __restrict__ x,
                                     unsigned short* __restrict__ xn, int V) {
  __shared__ float red[256];
  __shared__ float sinv;
  int m = blockIdx.x;
  const float* row = x + (size_t)m * V;
  float s = 0.f;
  for (int v = threadIdx.x; v < V; v += 256) { float t = row[v]; s += t * t; }
  red[threadIdx.x] = s; __syncthreads();
  for (int k = 128; k > 0; k >>= 1) {
    if (threadIdx.x < (unsigned)k) red[threadIdx.x] += red[threadIdx.x + k];
    __syncthreads();
  }
  if (threadIdx.x == 0) sinv = rsqrtf(red[0] / (float)V + EPSF);
  __syncthreads();
  float inv = sinv;
  for (int v = threadIdx.x; v < V; v += 256)
    xn[(size_t)m * V + v] = f2bf(row[v] * inv);
}

// depthwise causal conv on rms-normed x; x1 = x + conv_scale*(conv + bias)
__global__ void conv_kernel(const float* __restrict__ x, const float* __restrict__ r1,
                            const float* __restrict__ cw, const float* __restrict__ cb,
                            const float* __restrict__ cs, float* __restrict__ x1) {
  int t = blockIdx.x, b = blockIdx.y;
  int v0 = threadIdx.x * 4;
  float4 acc = make_float4(0.f, 0.f, 0.f, 0.f);
  for (int k = 0; k < 7; ++k) {
    int tp = t - 6 + k;
    if (tp < 0) continue;
    size_t rb = ((size_t)b * CT + tp) * CV;
    float inv = r1[b * CT + tp];
    float4 xv = *(const float4*)(x + rb + v0);
    acc.x += xv.x * inv * cw[(v0 + 0) * 7 + k];
    acc.y += xv.y * inv * cw[(v0 + 1) * 7 + k];
    acc.z += xv.z * inv * cw[(v0 + 2) * 7 + k];
    acc.w += xv.w * inv * cw[(v0 + 3) * 7 + k];
  }
  size_t rowb = ((size_t)b * CT + t) * CV;
  float4 xc = *(const float4*)(x + rowb + v0);
  float4 o;
  o.x = xc.x + cs[v0 + 0] * (acc.x + cb[v0 + 0]);
  o.y = xc.y + cs[v0 + 1] * (acc.y + cb[v0 + 1]);
  o.z = xc.z + cs[v0 + 2] * (acc.z + cb[v0 + 2]);
  o.w = xc.w + cs[v0 + 3] * (acc.w + cb[v0 + 3]);
  *(float4*)(x1 + rowb + v0) = o;
}

// gather g, its rms-norm (bf16) and per-step log-decay
__global__ void gather_kernel(const float* __restrict__ x1, const int* __restrict__ ridx,
                              const float* __restrict__ dw, const float* __restrict__ db,
                              float* __restrict__ g, unsigned short* __restrict__ gn,
                              float* __restrict__ ldb) {
  __shared__ float red[256];
  __shared__ float sinv;
  int t = blockIdx.x, b = blockIdx.y;
  int c = threadIdx.x;
  size_t m = (size_t)b * CT + t;
  float val = x1[m * CV + ridx[c]];
  red[c] = val * val; __syncthreads();
  for (int k = 128; k > 0; k >>= 1) {
    if (c < k) red[c] += red[c + k];
    __syncthreads();
  }
  if (c == 0) sinv = rsqrtf(red[0] / (float)CKA + EPSF);
  __syncthreads();
  float gnv = val * sinv;
  g[m * CKA + c]  = val;
  gn[m * CKA + c] = f2bf(gnv);
  red[c] = gnv * dw[c]; __syncthreads();
  for (int k = 128; k > 0; k >>= 1) {
    if (c < k) red[c] += red[c + k];
    __syncthreads();
  }
  if (c == 0) {
    float d = 1.f / (1.f + __expf(-(red[0] + db[0])));
    ldb[m] = logf(d + 1e-8f);
  }
}

__global__ void cvt_bf16_kernel(const float* __restrict__ s,
                                unsigned short* __restrict__ d, int n) {
  int i = blockIdx.x * 256 + threadIdx.x;
  if (i < n) d[i] = f2bf(s[i]);
}

// column-softmax of basis @ read_coeffs.T : softw[v][c], one block per column c
__global__ void dct_softmax_kernel(const float* __restrict__ basis,
                                   const float* __restrict__ rc,
                                   float* __restrict__ softw) {
  __shared__ float red[256];
  __shared__ float sval;
  int c = blockIdx.x;
  float lv[4];
  float lmax = -1e30f;
  for (int i = 0; i < 4; ++i) {
    int v = threadIdx.x + i * 256;
    float acc = 0.f;
    for (int j = 0; j < CNB; ++j) acc += basis[v * CNB + j] * rc[c * CNB + j];
    lv[i] = acc; lmax = fmaxf(lmax, acc);
  }
  red[threadIdx.x] = lmax; __syncthreads();
  for (int k = 128; k > 0; k >>= 1) {
    if (threadIdx.x < (unsigned)k)
      red[threadIdx.x] = fmaxf(red[threadIdx.x], red[threadIdx.x + k]);
    __syncthreads();
  }
  if (threadIdx.x == 0) sval = red[0];
  __syncthreads();
  float mx = sval, ssum = 0.f;
  for (int i = 0; i < 4; ++i) { lv[i] = __expf(lv[i] - mx); ssum += lv[i]; }
  __syncthreads();
  red[threadIdx.x] = ssum; __syncthreads();
  for (int k = 128; k > 0; k >>= 1) {
    if (threadIdx.x < (unsigned)k) red[threadIdx.x] += red[threadIdx.x + k];
    __syncthreads();
  }
  if (threadIdx.x == 0) sval = red[0];
  __syncthreads();
  float tot = sval;
  for (int i = 0; i < 4; ++i)
    softw[(size_t)(threadIdx.x + i * 256) * CNC + c] = lv[i] / tot;
}

// RWT[n][v] = (softw @ channel_mix)[v][n], bf16, transposed for B-fragments
__global__ void dct_rwt_kernel(const float* __restrict__ softw,
                               const float* __restrict__ cm,
                               unsigned short* __restrict__ rwt) {
  int id = blockIdx.x * 256 + threadIdx.x;  // NC*V ids
  int n = id / CV, v = id % CV;
  float acc = 0.f;
  for (int c = 0; c < CNC; ++c) acc += softw[(size_t)v * CNC + c] * cm[c * CNC + n];
  rwt[(size_t)n * CV + v] = f2bf(acc);
}

// WW[v][n] = basis @ write_coeffs.T, bf16, [V,NC] row-major
__global__ void dct_ww_kernel(const float* __restrict__ basis,
                              const float* __restrict__ wc,
                              unsigned short* __restrict__ ww) {
  int id = blockIdx.x * 256 + threadIdx.x;  // V*NC ids
  int v = id / CNC, n = id % CNC;
  float acc = 0.f;
  for (int j = 0; j < CNB; ++j) acc += basis[v * CNB + j] * wc[n * CNB + j];
  ww[(size_t)v * CNC + n] = f2bf(acc);
}

// bf16 [T,D] -> [D,T] transpose per batch
__global__ void transpose_kernel(const unsigned short* __restrict__ v,
                                 unsigned short* __restrict__ vT) {
  __shared__ unsigned short tile[32][33];
  int b = blockIdx.z;
  int t0 = blockIdx.x * 32, d0 = blockIdx.y * 32;
  int tx = threadIdx.x & 31, ty = threadIdx.x >> 5;  // 32x8
  const unsigned short* src = v + (size_t)b * CT * CKA;
  for (int i = ty; i < 32; i += 8)
    tile[i][tx] = src[(size_t)(t0 + i) * CKA + d0 + tx];
  __syncthreads();
  unsigned short* dst = vT + (size_t)b * CKA * CT;
  for (int i = ty; i < 32; i += 8)
    dst[(size_t)(d0 + i) * CT + t0 + tx] = tile[tx][i];
}

// ---------------- WMMA GEMM kernels ----------------

// q/k/v = gn @ W^T, bf16 in / bf16 out, fp32 accumulate.
// grid (M/16, 6), 8 waves/block; slot = matrix*16 + n-tile.
__global__ void __launch_bounds__(256) qkv_kernel(
    const unsigned short* __restrict__ gn,
    const unsigned short* __restrict__ Wqb, const unsigned short* __restrict__ Wkb,
    const unsigned short* __restrict__ Wvb,
    unsigned short* __restrict__ q, unsigned short* __restrict__ k,
    unsigned short* __restrict__ v) {
  int lane = threadIdx.x & 31, wid = threadIdx.x >> 5;
  int m0 = blockIdx.x * 16;
  int slot = blockIdx.y * 8 + wid;           // 0..47
  int mat = slot >> 4, nt = slot & 15;
  const unsigned short* W = (mat == 0) ? Wqb : (mat == 1) ? Wkb : Wvb;
  unsigned short* out = (mat == 0) ? q : (mat == 1) ? k : v;
  const unsigned short* abase = gn + (size_t)m0 * CKA;
  const unsigned short* brow  = W + (size_t)(nt * 16 + (lane & 15)) * CKA;
  v8f acc = {};
  for (int k0 = 0; k0 < CKA; k0 += 32) {
    v16bf a = load_a_frag(abase, CKA, k0, lane);
    v16bf b = load_b_frag(brow, k0, lane);
    acc = wmma_bf16(a, b, acc);
  }
  int ncol = nt * 16 + (lane & 15);
  int rb = (lane & 16) ? 8 : 0;
  for (int r = 0; r < 8; ++r)
    out[(size_t)(m0 + rb + r) * CKA + ncol] = f2bf(acc[r]);
}

// DCT apply: vals = gelu(xn2 @ RW + bias); xout = x1 + (vals @ WW^T) * scale
// grid M/16, 8 waves/block.
__global__ void __launch_bounds__(256) dct_apply_kernel(
    const unsigned short* __restrict__ xn2, const unsigned short* __restrict__ rwt,
    const unsigned short* __restrict__ ww, const float* __restrict__ bias,
    const float* __restrict__ x1, float* __restrict__ xout,
    const float* __restrict__ scale) {
  __shared__ __align__(16) unsigned short vals[16 * CNC];
  int lane = threadIdx.x & 31, wid = threadIdx.x >> 5;
  int m0 = blockIdx.x * 16;
  {  // phase 1: vals n-slice per wave (NC = 8 waves * 16)
    int c0 = wid * 16;
    const unsigned short* abase = xn2 + (size_t)m0 * CV;
    const unsigned short* brow  = rwt + (size_t)(c0 + (lane & 15)) * CV;
    v8f acc = {};
    for (int k0 = 0; k0 < CV; k0 += 32) {
      v16bf a = load_a_frag(abase, CV, k0, lane);
      v16bf b = load_b_frag(brow, k0, lane);
      acc = wmma_bf16(a, b, acc);
    }
    int ccol = c0 + (lane & 15);
    float bv = bias[ccol];
    int rb = (lane & 16) ? 8 : 0;
    for (int r = 0; r < 8; ++r) {
      float x = acc[r] + bv;
      float gel = 0.5f * x * (1.f + erff(x * 0.70710678118654752f));
      vals[(rb + r) * CNC + ccol] = f2bf(gel);
    }
  }
  __syncthreads();
  float ds = scale[0];
  for (int nt = 0; nt < 8; ++nt) {  // phase 2: 8 v-tiles per wave
    int v0 = wid * 128 + nt * 16;
    const unsigned short* brow = ww + (size_t)(v0 + (lane & 15)) * CNC;
    v8f acc = {};
    for (int k0 = 0; k0 < CNC; k0 += 32) {
      v16bf a = load_a_frag(vals, CNC, k0, lane);
      v16bf b = load_b_frag(brow, k0, lane);
      acc = wmma_bf16(a, b, acc);
    }
    int col = v0 + (lane & 15);
    int rb = (lane & 16) ? 8 : 0;
    for (int r = 0; r < 8; ++r) {
      size_t idx = (size_t)(m0 + rb + r) * CV + col;
      xout[idx] = x1[idx] + ds * acc[r];
    }
  }
}

// Flash-style anti-causal decay attention + Wo + row-rms + scatter-add.
// 16 waves / 16 query rows per block; grid (T/16, B).
// q A-fragments are chunk-loop invariant -> cached in registers (8 x v16bf);
// next chunk's k / vT rows are software-prefetched (global_prefetch_b8).
__global__ void __launch_bounds__(512) attn_kernel(
    const unsigned short* __restrict__ q, const unsigned short* __restrict__ k,
    const unsigned short* __restrict__ vT, const unsigned short* __restrict__ Wob,
    const float* __restrict__ g, const float* __restrict__ ldb,
    const int* __restrict__ widx, float* __restrict__ xout,
    const float* __restrict__ mem_out_scale, const float* __restrict__ mem_scale,
    const float* __restrict__ write_scale) {
  __shared__ __align__(16) unsigned short s_tile[16 * 256];  // scores / retrieved (bf16)
  __shared__ float gout[16 * 256];
  __shared__ float invr[16];
  int lane = threadIdx.x & 31, wid = threadIdx.x >> 5;  // 16 waves
  int i0 = blockIdx.x * 16;
  int b = blockIdx.y;
  size_t mb = (size_t)b * CT;
  int n0 = wid * 16;

  // register-cache the q tile fragments (invariant across key chunks)
  v16bf qfrag[8];
  {
    const unsigned short* abase = q + (mb + i0) * CKA;
#pragma unroll
    for (int kk = 0; kk < 8; ++kk)
      qfrag[kk] = load_a_frag(abase, CKA, kk * 32, lane);
  }
  const unsigned short* vbase = vT + ((size_t)b * CKA + n0 + (lane & 15)) * CT;

  v8f R = {};
  for (int jc = i0 >> 8; jc < CT / 256; ++jc) {
    int jbase = jc * 256;
    // prefetch next chunk's k rows (this wave's 16 keys) and vT span
    if (jbase + 256 < CT) {
      __builtin_prefetch(k + (mb + jbase + 256 + wid * 16 + (lane & 15)) * CKA, 0, 1);
      __builtin_prefetch(vbase + jbase + 256, 0, 1);
    }
    __syncthreads();  // previous chunk's reads done before overwrite
    {  // phase A: this wave computes S for keys [jbase+wid*16, +16)
      int j0 = jbase + wid * 16;
      const unsigned short* brow = k + (mb + j0 + (lane & 15)) * CKA;
      v8f s = {};
#pragma unroll
      for (int kk = 0; kk < 8; ++kk) {
        v16bf bb = load_b_frag(brow, kk * 32, lane);
        s = wmma_bf16(qfrag[kk], bb, s);
      }
      int j = j0 + (lane & 15);
      float ldj = ldb[mb + j];
      int rb = (lane & 16) ? 8 : 0;
      for (int r = 0; r < 8; ++r) {
        int i = i0 + rb + r;
        float w = 0.f;
        if (j > i) w = __expf(ldj * (float)(j - i - 1));
        s_tile[(rb + r) * 256 + (j - jbase)] = f2bf(s[r] * 0.0625f * w);
      }
    }
    __syncthreads();
    {  // phase B: R += S_w @ v  (this wave owns d-slice n0)
      const unsigned short* vrow = vbase + jbase;
      for (int k0 = 0; k0 < 256; k0 += 32) {
        v16bf a = load_a_frag(s_tile, 256, k0, lane);
        v16bf bb = load_b_frag(vrow, k0, lane);
        R = wmma_bf16(a, bb, R);
      }
    }
  }
  __syncthreads();
  {  // stage retrieved tile (bf16) for the Wo matmul
    int rb = (lane & 16) ? 8 : 0;
    for (int r = 0; r < 8; ++r)
      s_tile[(rb + r) * 256 + n0 + (lane & 15)] = f2bf(R[r]);
  }
  __syncthreads();
  float mos = mem_out_scale[0], ms = mem_scale[0];
  {  // mem = retrieved @ Wo^T ; gout = g + ms*mos*mem
    const unsigned short* wrow = Wob + (size_t)(n0 + (lane & 15)) * CKA;
    v8f macc = {};
    for (int k0 = 0; k0 < CKA; k0 += 32) {
      v16bf a = load_a_frag(s_tile, 256, k0, lane);
      v16bf bb = load_b_frag(wrow, k0, lane);
      macc = wmma_bf16(a, bb, macc);
    }
    int col = n0 + (lane & 15);
    int rb = (lane & 16) ? 8 : 0;
    for (int r = 0; r < 8; ++r) {
      int row = i0 + rb + r;
      float gv = g[(mb + row) * CKA + col];
      gout[(rb + r) * 256 + col] = gv + ms * mos * macc[r];
    }
  }
  __syncthreads();
  if (threadIdx.x < 16) {  // per-row inv rms over 256
    float s = 0.f;
    for (int c = 0; c < 256; ++c) { float t = gout[threadIdx.x * 256 + c]; s += t * t; }
    invr[threadIdx.x] = rsqrtf(s / 256.f + EPSF);
  }
  __syncthreads();
  float wsc = write_scale[0] * 0.0625f;  // / sqrt(256)
  for (int it = 0; it < 8; ++it) {       // 16*256 elems / 512 threads
    int idx = threadIdx.x + it * 512;
    int r = idx >> 8, c = idx & 255;
    float val = gout[idx] * invr[r] * wsc;
    size_t o = (mb + i0 + r) * CV + widx[c];
    xout[o] += val;  // block owns these rows exclusively; widx unique
  }
}

// ---------------- launcher ----------------
extern "C" void kernel_launch(void* const* d_in, const int* in_sizes, int n_in,
                              void* d_out, int out_size, void* d_ws, size_t ws_size,
                              hipStream_t stream) {
  (void)in_sizes; (void)n_in; (void)out_size; (void)ws_size;
  const float* x            = (const float*)d_in[0];
  const float* basis        = (const float*)d_in[1];
  const float* conv_w       = (const float*)d_in[2];
  const float* conv_b       = (const float*)d_in[3];
  const float* Wq           = (const float*)d_in[4];
  const float* Wk           = (const float*)d_in[5];
  const float* Wv           = (const float*)d_in[6];
  const float* Wo           = (const float*)d_in[7];
  const float* decay_w      = (const float*)d_in[8];
  const float* decay_b      = (const float*)d_in[9];
  const float* mem_out_sc   = (const float*)d_in[10];
  const float* read_coeffs  = (const float*)d_in[11];
  const float* write_coeffs = (const float*)d_in[12];
  const float* channel_mix  = (const float*)d_in[13];
  const float* dct_bias     = (const float*)d_in[14];
  const float* dct_out_sc   = (const float*)d_in[15];
  const float* conv_scale   = (const float*)d_in[16];
  const float* mem_scale    = (const float*)d_in[17];
  const float* write_scale  = (const float*)d_in[18];
  const int*   read_idx     = (const int*)d_in[19];
  const int*   write_idx    = (const int*)d_in[20];
  float* xout = (float*)d_out;

  size_t off = 0;
  auto alloc = [&](size_t bytes) {
    void* p = (char*)d_ws + off;
    off += (bytes + 255) & ~(size_t)255;
    return p;
  };
  float*          x1    = (float*)alloc((size_t)CM * CV * 4);
  float*          r1    = (float*)alloc((size_t)CM * 4);
  unsigned short* xn2   = (unsigned short*)alloc((size_t)CM * CV * 2);
  float*          g     = (float*)alloc((size_t)CM * CKA * 4);
  unsigned short* gn    = (unsigned short*)alloc((size_t)CM * CKA * 2);
  float*          ldb   = (float*)alloc((size_t)CM * 4);
  unsigned short* qb    = (unsigned short*)alloc((size_t)CM * CKA * 2);
  unsigned short* kb    = (unsigned short*)alloc((size_t)CM * CKA * 2);
  unsigned short* vb    = (unsigned short*)alloc((size_t)CM * CKA * 2);
  unsigned short* vTb   = (unsigned short*)alloc((size_t)CM * CKA * 2);
  unsigned short* Wqb   = (unsigned short*)alloc((size_t)CKA * CKA * 2);
  unsigned short* Wkb   = (unsigned short*)alloc((size_t)CKA * CKA * 2);
  unsigned short* Wvb   = (unsigned short*)alloc((size_t)CKA * CKA * 2);
  unsigned short* Wob   = (unsigned short*)alloc((size_t)CKA * CKA * 2);
  float*          softw = (float*)alloc((size_t)CV * CNC * 4);
  unsigned short* rwt   = (unsigned short*)alloc((size_t)CNC * CV * 2);
  unsigned short* ww    = (unsigned short*)alloc((size_t)CV * CNC * 2);

  // 1. rms of raw x
  rms_rowinv_kernel<<<CM, 256, 0, stream>>>(x, r1, CV);
  // 2. causal depthwise conv -> x1
  conv_kernel<<<dim3(CT, CB), 256, 0, stream>>>(x, r1, conv_w, conv_b, conv_scale, x1);
  // 3. rms-norm x1 -> bf16 (for DCT)
  rms_norm_bf16_kernel<<<CM, 256, 0, stream>>>(x1, xn2, CV);
  // 4. gather registers, norm, log-decay
  gather_kernel<<<dim3(CT, CB), 256, 0, stream>>>(x1, read_idx, decay_w, decay_b,
                                                  g, gn, ldb);
  // 5. weight conversions
  cvt_bf16_kernel<<<CKA * CKA / 256, 256, 0, stream>>>(Wq, Wqb, CKA * CKA);
  cvt_bf16_kernel<<<CKA * CKA / 256, 256, 0, stream>>>(Wk, Wkb, CKA * CKA);
  cvt_bf16_kernel<<<CKA * CKA / 256, 256, 0, stream>>>(Wv, Wvb, CKA * CKA);
  cvt_bf16_kernel<<<CKA * CKA / 256, 256, 0, stream>>>(Wo, Wob, CKA * CKA);
  // 6. q/k/v projections (WMMA)
  qkv_kernel<<<dim3(CM / 16, 6), 256, 0, stream>>>(gn, Wqb, Wkb, Wvb, qb, kb, vb);
  // 7. v -> vT for B-fragment contiguity
  transpose_kernel<<<dim3(CT / 32, CKA / 32, CB), 256, 0, stream>>>(vb, vTb);
  // 8. DCT weight precompute
  dct_softmax_kernel<<<CNC, 256, 0, stream>>>(basis, read_coeffs, softw);
  dct_rwt_kernel<<<CNC * CV / 256, 256, 0, stream>>>(softw, channel_mix, rwt);
  dct_ww_kernel<<<CV * CNC / 256, 256, 0, stream>>>(basis, write_coeffs, ww);
  // 9. DCT apply -> xout = x1 + delta (WMMA)
  dct_apply_kernel<<<CM / 16, 256, 0, stream>>>(xn2, rwt, ww, dct_bias, x1, xout,
                                                dct_out_sc);
  // 10. decay attention + Wo + rms + scatter-add into xout (WMMA)
  attn_kernel<<<dim3(CT / 16, CB), 512, 0, stream>>>(qb, kb, vTb, Wob, g, ldb,
                                                     write_idx, xout, mem_out_sc,
                                                     mem_scale, write_scale);
}
